// TinyMoE_18923625906579
// MI455X (gfx1250) — matrix operand order repaired and checked by reading, not compiled, not run on hardware
//
#include <hip/hip_runtime.h>

// ---------------- problem constants ----------------
static constexpr int HDIM  = 1024;   // hidden size
static constexpr int IDIM  = 1024;   // moe intermediate size
static constexpr int NE    = 8;      // routed experts
static constexpr int TT    = 4096;   // tokens = B*S
static constexpr int NSLOT = 2 * TT; // token-expert slots (K=2)

// ---------------- CDNA5 WMMA types ----------------
typedef __attribute__((ext_vector_type(16))) __bf16       v16bf;
typedef __attribute__((ext_vector_type(8)))  float        v8f;
typedef __attribute__((ext_vector_type(4)))  unsigned int v4u;

union Frag { v16bf v; v4u u[2]; };

__device__ __forceinline__ unsigned short f2bf(float f) {
  unsigned u = __float_as_uint(f);
  u += 0x7FFFu + ((u >> 16) & 1u);          // round-to-nearest-even
  return (unsigned short)(u >> 16);
}
__device__ __forceinline__ unsigned pack2(float lo, float hi) {
  return (unsigned)f2bf(lo) | ((unsigned)f2bf(hi) << 16);
}
__device__ __forceinline__ v8f wmma_bf16(const Frag& a, const Frag& b, v8f c) {
  return __builtin_amdgcn_wmma_f32_16x16x32_bf16(false, a.v, false, b.v,
                                                 (short)0, c, false, false);
}

// ------- CDNA5 async global->LDS copy (ASYNCcnt-tracked DMA, no VGPR data) ----
__device__ __forceinline__ void async_ld_b128(unsigned ldsByte, const void* g) {
  asm volatile("global_load_async_to_lds_b128 %0, %1, off"
               :: "v"(ldsByte), "v"(g) : "memory");
}
__device__ __forceinline__ void wait_async0() {
#if __has_builtin(__builtin_amdgcn_s_wait_asynccnt)
  __builtin_amdgcn_s_wait_asynccnt(0);
#else
  asm volatile("s_wait_asynccnt 0x0" ::: "memory");
#endif
}
// low 32 bits of a flat shared pointer == LDS byte address (ISA aperture rule)
__device__ __forceinline__ unsigned lds_addr(const void* p) {
  return (unsigned)(unsigned long long)p;
}

// ---------------- fp32 -> bf16 activation convert ----------------
__global__ void k_cvt_x(const float* __restrict__ x,
                        unsigned short* __restrict__ xb, int n8) {
  int i = blockIdx.x * 256 + threadIdx.x;
  if (i >= n8) return;
  const float4* p = (const float4*)x;
  float4 a = p[2 * i], b = p[2 * i + 1];
  v4u v = { pack2(a.x, a.y), pack2(a.z, a.w), pack2(b.x, b.y), pack2(b.z, b.w) };
  ((v4u*)xb)[i] = v;
}

// ------- fp32 weights -> bf16, pre-swizzled to B-fragment-linear order -------
// For each 32(K)x16(N) tile: 32 lanes x 16 bf16; lane L holds column n=L&15,
// k = kt*32 + (L>>4)*16 + e  (contiguous k order, same as A in LDS). Tiles stored kt-major
// inside each column-block nt:  flat_run = (nt*nK + kt)*32 + L.
__global__ void k_swz(const float* __restrict__ src,
                      unsigned short* __restrict__ dst,
                      int K, int N, int nMat) {
  long run = (long)blockIdx.x * 256 + threadIdx.x;
  long runsPerMat = (long)K * N / 16;
  if (run >= runsPerMat * nMat) return;
  int  mat = (int)(run / runsPerMat);
  long r2  = run % runsPerMat;
  int  nK  = K / 32;
  int  nt  = (int)(r2 / (nK * 32));
  int  r3  = (int)(r2 % (nK * 32));
  int  kt  = r3 >> 5;
  int  L   = r3 & 31;
  int  n     = nt * 16 + (L & 15);
  int  kbase = kt * 32 + ((L >> 4) * 16);
  const float* s = src + (size_t)mat * K * N;
  unsigned w[8];
#pragma unroll
  for (int e2 = 0; e2 < 8; ++e2) {
    float lo = s[(size_t)(kbase + 2 * e2)     * N + n];
    float hi = s[(size_t)(kbase + 2 * e2 + 1) * N + n];
    w[e2] = pack2(lo, hi);
  }
  v4u* dp = (v4u*)(dst + (size_t)run * 16);
  v4u v0 = { w[0], w[1], w[2], w[3] };
  v4u v1 = { w[4], w[5], w[6], w[7] };
  dp[0] = v0; dp[1] = v1;
}

// ---------------- router: fp32 logits, softmax, top-2 ----------------
__global__ __launch_bounds__(128) void k_router(
    const float* __restrict__ x, const float* __restrict__ rw,
    int* __restrict__ tIdx, float* __restrict__ tW, int* __restrict__ meta) {
  int t    = blockIdx.x * 4 + (threadIdx.x >> 5);
  int lane = threadIdx.x & 31;
  const float* xr = x + (size_t)t * HDIM;
  float a[NE];
#pragma unroll
  for (int e = 0; e < NE; ++e) a[e] = 0.f;
  for (int j = 0; j < HDIM / 32; ++j) {
    float xv = xr[j * 32 + lane];
#pragma unroll
    for (int e = 0; e < NE; ++e) a[e] += xv * rw[e * HDIM + j * 32 + lane];
  }
#pragma unroll
  for (int e = 0; e < NE; ++e) {
    float v = a[e];
    for (int off = 16; off > 0; off >>= 1) v += __shfl_xor(v, off, 32);
    a[e] = v;
  }
  if (lane == 0) {
    float m = a[0];
#pragma unroll
    for (int e = 1; e < NE; ++e) m = fmaxf(m, a[e]);
    float p[NE], s = 0.f;
#pragma unroll
    for (int e = 0; e < NE; ++e) { p[e] = __expf(a[e] - m); s += p[e]; }
    float inv = 1.f / s;
    int i0 = 0; float b0 = p[0];
#pragma unroll
    for (int e = 1; e < NE; ++e) if (p[e] > b0) { b0 = p[e]; i0 = e; }
    int i1 = -1; float b1 = -1.f;
#pragma unroll
    for (int e = 0; e < NE; ++e) if (e != i0 && p[e] > b1) { b1 = p[e]; i1 = e; }
    tIdx[t]       = i0 | (i1 << 8);
    tW[2 * t]     = b0 * inv;     // routed_scaling_factor = 1.0
    tW[2 * t + 1] = b1 * inv;
    atomicAdd(&meta[i0], 1);
    atomicAdd(&meta[i1], 1);
  }
}

// meta: [0..7]=cnt  [8..15]=cursor  [16..24]=offsets
__global__ void k_offsets(int* meta) {
  if (threadIdx.x == 0) {
    int acc = 0;
    for (int e = 0; e < NE; ++e) { meta[16 + e] = acc; acc += meta[e]; }
    meta[24] = acc;
  }
}

__global__ void k_assign(const int* __restrict__ tIdx,
                         const float* __restrict__ tW, int* __restrict__ meta,
                         int* __restrict__ slotDest, float* __restrict__ slotW) {
  int t = blockIdx.x * 256 + threadIdx.x;
  if (t >= TT) return;
  int pk = tIdx[t];
#pragma unroll
  for (int k = 0; k < 2; ++k) {
    int e   = (k == 0) ? (pk & 255) : ((pk >> 8) & 255);
    int pos = meta[16 + e] + atomicAdd(&meta[8 + e], 1);
    slotDest[pos] = t * 2 + k;        // token = dest>>1
    slotW[pos]    = tW[t * 2 + k];
  }
}

// --------- helper: locate expert tile for routed GEMMs ---------
struct TileInfo { int cntLoc; int slotBase; int expert; };
__device__ __forceinline__ bool find_tile(const int* meta, int mt, TileInfo& ti) {
  for (int e = 0; e < NE; ++e) {
    int tiles = (meta[e] + 127) >> 7;
    if (mt < tiles) {
      int cnt   = meta[e];
      ti.expert = e;
      ti.slotBase = meta[16 + e] + mt * 128;
      ti.cntLoc = cnt - mt * 128;
      if (ti.cntLoc > 128) ti.cntLoc = 128;
      return true;
    }
    mt -= tiles;
  }
  return false;
}

// ---------------- fused gate+up GEMM: hid = sigmoid(x*Wg) .* (x*Wu) ----------
// BM=128, BN=64, 4 waves. Double-buffered LDS A-stage fed by async global->LDS
// DMA; B fragments loaded directly from swizzled weights with a 2-deep
// register pipeline.
template <bool ROUTED>
__global__ __launch_bounds__(128) void k_gateup(
    const unsigned short* __restrict__ xb, const unsigned short* __restrict__ wg,
    const unsigned short* __restrict__ wu, unsigned short* __restrict__ hid,
    const int* __restrict__ meta, const int* __restrict__ slotDest) {
  __shared__ unsigned short lA[2][128 * 32];
  const int tid = threadIdx.x, wave = tid >> 5, L = tid & 31;
  const int nBlk = blockIdx.y;

  int rowBase = 0, cntLoc = 128, slotBase = 0;
  const unsigned short *pg = wg, *pu = wu;
  if constexpr (ROUTED) {
    TileInfo ti;
    if (!find_tile(meta, blockIdx.x, ti)) return;
    cntLoc = ti.cntLoc; slotBase = ti.slotBase;
    pg = wg + (size_t)ti.expert * HDIM * IDIM;
    pu = wu + (size_t)ti.expert * HDIM * IDIM;
  } else {
    rowBase = blockIdx.x * 128;
  }

  // k-invariant staging addressing: 4 chunks per thread
  const unsigned short* gsrc[4];
  bool     gval[4];
  int      ldsOff[4];
  unsigned ldsByte[4];
#pragma unroll
  for (int i = 0; i < 4; ++i) {
    int p = tid + 128 * i, r = p >> 2, c = p & 3;
    bool valid = (!ROUTED) || (r < cntLoc);
    int  row   = 0;
    if (valid) row = ROUTED ? (slotDest[slotBase + r] >> 1) : (rowBase + r);
    gsrc[i]    = xb + (size_t)row * HDIM + c * 8;
    gval[i]    = valid;
    ldsOff[i]  = (r >> 4) * 512 + ((r & 15) + 16 * (c >> 1)) * 16 + (c & 1) * 8;
    ldsByte[i] = lds_addr(&lA[0][ldsOff[i]]);
  }

  v8f accG[2][4], accU[2][4];
#pragma unroll
  for (int a = 0; a < 2; ++a)
#pragma unroll
    for (int b = 0; b < 4; ++b) { accG[a][b] = {}; accU[a][b] = {}; }

  const int nK = HDIM / 32;
  const size_t niStep = (size_t)nK * 32;   // 16-ushort runs per nt step

  // prologue: zero pad slots once (both buffers), async-stage kt=0 into buf 0
#pragma unroll
  for (int i = 0; i < 4; ++i) {
    if (gval[i]) {
      async_ld_b128(ldsByte[i], gsrc[i]);
    } else {
      v4u z = { 0, 0, 0, 0 };
      *(v4u*)(&lA[0][ldsOff[i]]) = z;
      *(v4u*)(&lA[1][ldsOff[i]]) = z;
    }
  }
  wait_async0();
  __syncthreads();

  for (int kt = 0; kt < nK; ++kt) {
    const int  cur = kt & 1;
    const bool hs  = (kt + 1 < nK);
    // async-stage next k-step into the other buffer (runs under the WMMAs)
    if (hs) {
#pragma unroll
      for (int i = 0; i < 4; ++i)
        if (gval[i])
          async_ld_b128(ldsByte[i] + (cur ^ 1) * 8192,
                        gsrc[i] + (size_t)(kt + 1) * 32);
    }
    // A fragments from current LDS buffer
    Frag a0, a1;
    { const v4u* ap = (const v4u*)(&lA[cur][(2 * wave) * 512 + L * 16]);
      a0.u[0] = ap[0]; a0.u[1] = ap[1]; }
    { const v4u* ap = (const v4u*)(&lA[cur][(2 * wave + 1) * 512 + L * 16]);
      a1.u[0] = ap[0]; a1.u[1] = ap[1]; }
    // B fragments: 2-deep register pipeline over the 4 N-tiles
    const size_t wbase = (((size_t)nBlk * 4) * nK + kt) * 32 + L;
    Frag bgN, buN;
    { const v4u* bp = (const v4u*)(pg + wbase * 16); bgN.u[0] = bp[0]; bgN.u[1] = bp[1]; }
    { const v4u* bp = (const v4u*)(pu + wbase * 16); buN.u[0] = bp[0]; buN.u[1] = bp[1]; }
#pragma unroll
    for (int ni = 0; ni < 4; ++ni) {
      Frag bgC = bgN, buC = buN;
      if (ni < 3) {
        const v4u* g2 = (const v4u*)(pg + (wbase + (ni + 1) * niStep) * 16);
        const v4u* u2 = (const v4u*)(pu + (wbase + (ni + 1) * niStep) * 16);
        bgN.u[0] = g2[0]; bgN.u[1] = g2[1];
        buN.u[0] = u2[0]; buN.u[1] = u2[1];
      }
      accG[0][ni] = wmma_bf16(a0, bgC, accG[0][ni]);
      accG[1][ni] = wmma_bf16(a1, bgC, accG[1][ni]);
      accU[0][ni] = wmma_bf16(a0, buC, accU[0][ni]);
      accU[1][ni] = wmma_bf16(a1, buC, accU[1][ni]);
    }
    wait_async0();     // staged buffer complete before it is published
    __syncthreads();
  }

  // epilogue: sigmoid(g)*u -> bf16, row-major [rows, IDIM]
  const int colBase = nBlk * 64 + (L & 15);
  const int rHalf   = (L >> 4) * 8;
#pragma unroll
  for (int mi = 0; mi < 2; ++mi)
#pragma unroll
    for (int ni = 0; ni < 4; ++ni)
#pragma unroll
      for (int r = 0; r < 8; ++r) {
        int lr = 32 * wave + 16 * mi + rHalf + r;
        if (lr < cntLoc) {
          float g = accG[mi][ni][r], u = accU[mi][ni][r];
          float h = u / (1.f + __expf(-g));
          int orow = ROUTED ? (slotBase + lr) : (rowBase + lr);
          hid[(size_t)orow * IDIM + colBase + 16 * ni] = f2bf(h);
        }
      }
}

// ---------------- down projection GEMM ----------------
// shared: out[t] = resid[t] + hid*Wd ; routed: sel[dest] = w * (hid*Wd)
template <bool ROUTED>
__global__ __launch_bounds__(128) void k_down(
    const unsigned short* __restrict__ hid, const unsigned short* __restrict__ wd,
    float* __restrict__ outp, const float* __restrict__ resid,
    const int* __restrict__ meta, const int* __restrict__ slotDest,
    const float* __restrict__ slotW) {
  __shared__ unsigned short lA[2][128 * 32];
  const int tid = threadIdx.x, wave = tid >> 5, L = tid & 31;
  const int nBlk = blockIdx.y;

  int rowBase = 0, cntLoc = 128, slotBase = 0;
  const unsigned short* pw = wd;
  if constexpr (ROUTED) {
    TileInfo ti;
    if (!find_tile(meta, blockIdx.x, ti)) return;
    cntLoc = ti.cntLoc; slotBase = ti.slotBase;
    pw = wd + (size_t)ti.expert * IDIM * HDIM;
  } else {
    rowBase = blockIdx.x * 128;
  }

  const unsigned short* gsrc[4];
  unsigned ldsByte[4];
#pragma unroll
  for (int i = 0; i < 4; ++i) {
    int p = tid + 128 * i, r = p >> 2, c = p & 3;
    int rr  = (r < cntLoc) ? r : 0;                  // clamp pad rows
    int row = ROUTED ? (slotBase + rr) : (rowBase + rr);
    gsrc[i]    = hid + (size_t)row * IDIM + c * 8;
    int off    = (r >> 4) * 512 + ((r & 15) + 16 * (c >> 1)) * 16 + (c & 1) * 8;
    ldsByte[i] = lds_addr(&lA[0][off]);
  }

  v8f acc[2][4];
#pragma unroll
  for (int a = 0; a < 2; ++a)
#pragma unroll
    for (int b = 0; b < 4; ++b) acc[a][b] = {};

  const int nK = IDIM / 32;
  const size_t niStep = (size_t)nK * 32;

#pragma unroll
  for (int i = 0; i < 4; ++i) async_ld_b128(ldsByte[i], gsrc[i]);
  wait_async0();
  __syncthreads();

  for (int kt = 0; kt < nK; ++kt) {
    const int  cur = kt & 1;
    const bool hs  = (kt + 1 < nK);
    if (hs) {
#pragma unroll
      for (int i = 0; i < 4; ++i)
        async_ld_b128(ldsByte[i] + (cur ^ 1) * 8192,
                      gsrc[i] + (size_t)(kt + 1) * 32);
    }
    Frag a0, a1;
    { const v4u* ap = (const v4u*)(&lA[cur][(2 * wave) * 512 + L * 16]);
      a0.u[0] = ap[0]; a0.u[1] = ap[1]; }
    { const v4u* ap = (const v4u*)(&lA[cur][(2 * wave + 1) * 512 + L * 16]);
      a1.u[0] = ap[0]; a1.u[1] = ap[1]; }
    // fully hoisted B fragments: one long clause, overlapped waits
    const size_t wbase = (((size_t)nBlk * 4) * nK + kt) * 32 + L;
    Frag bd[4];
#pragma unroll
    for (int ni = 0; ni < 4; ++ni) {
      const v4u* bp = (const v4u*)(pw + (wbase + ni * niStep) * 16);
      bd[ni].u[0] = bp[0]; bd[ni].u[1] = bp[1];
    }
#pragma unroll
    for (int ni = 0; ni < 4; ++ni) {
      acc[0][ni] = wmma_bf16(a0, bd[ni], acc[0][ni]);
      acc[1][ni] = wmma_bf16(a1, bd[ni], acc[1][ni]);
    }
    wait_async0();
    __syncthreads();
  }

  const int rHalf = (L >> 4) * 8;
#pragma unroll
  for (int mi = 0; mi < 2; ++mi)
#pragma unroll
    for (int ni = 0; ni < 4; ++ni)
#pragma unroll
      for (int r = 0; r < 8; ++r) {
        int lr = 32 * wave + 16 * mi + rHalf + r;
        if (lr < cntLoc) {
          int col = nBlk * 64 + 16 * ni + (L & 15);
          if constexpr (ROUTED) {
            int   s    = slotBase + lr;
            int   dest = slotDest[s];
            float w    = slotW[s];
            outp[(size_t)dest * HDIM + col] = w * acc[mi][ni][r];
          } else {
            size_t idx = (size_t)(rowBase + lr) * HDIM + col;
            outp[idx] = resid[idx] + acc[mi][ni][r];
          }
        }
      }
}

// ---------------- final combine: out += sel[2t] + sel[2t+1] ----------------
__global__ void k_combine(float* __restrict__ out, const float* __restrict__ sel) {
  int i = blockIdx.x * 256 + threadIdx.x;           // over TT*HDIM/4
  if (i >= TT * (HDIM / 4)) return;
  const int HV = HDIM / 4;
  int t = i / HV, c = i % HV;
  float4*       o4 = (float4*)out;
  const float4* s4 = (const float4*)sel;
  float4 o  = o4[i];
  float4 s0 = s4[(size_t)(2 * t) * HV + c];
  float4 s1 = s4[(size_t)(2 * t + 1) * HV + c];
  o.x += s0.x + s1.x; o.y += s0.y + s1.y;
  o.z += s0.z + s1.z; o.w += s0.w + s1.w;
  o4[i] = o;
}

// ---------------- host launch ----------------
extern "C" void kernel_launch(void* const* d_in, const int* in_sizes, int n_in,
                              void* d_out, int out_size, void* d_ws, size_t ws_size,
                              hipStream_t stream) {
  const float* x   = (const float*)d_in[0];
  const float* rw  = (const float*)d_in[1];
  const float* shg = (const float*)d_in[2];
  const float* shu = (const float*)d_in[3];
  const float* shd = (const float*)d_in[4];
  const float* exg = (const float*)d_in[5];
  const float* exu = (const float*)d_in[6];
  const float* exd = (const float*)d_in[7];
  float* out = (float*)d_out;

  char* w = (char*)d_ws;
  auto alloc = [&](size_t bytes) {
    char* p = w; w += (bytes + 255) & ~(size_t)255; return p;
  };
  unsigned short* xb   = (unsigned short*)alloc((size_t)TT * HDIM * 2);
  unsigned short* wgs  = (unsigned short*)alloc((size_t)HDIM * IDIM * 2);
  unsigned short* wus  = (unsigned short*)alloc((size_t)HDIM * IDIM * 2);
  unsigned short* wds  = (unsigned short*)alloc((size_t)IDIM * HDIM * 2);
  unsigned short* wge  = (unsigned short*)alloc((size_t)NE * HDIM * IDIM * 2);
  unsigned short* wue  = (unsigned short*)alloc((size_t)NE * HDIM * IDIM * 2);
  unsigned short* wde  = (unsigned short*)alloc((size_t)NE * IDIM * HDIM * 2);
  unsigned short* hidS = (unsigned short*)alloc((size_t)TT * IDIM * 2);
  unsigned short* hidR = (unsigned short*)alloc((size_t)NSLOT * IDIM * 2);
  float*          sel  = (float*)alloc((size_t)NSLOT * HDIM * 4);
  int*            meta = (int*)alloc(32 * 4);
  int*            tIdx = (int*)alloc((size_t)TT * 4);
  float*          tW   = (float*)alloc((size_t)NSLOT * 4);
  int*          slotDest = (int*)alloc((size_t)NSLOT * 4);
  float*        slotW    = (float*)alloc((size_t)NSLOT * 4);

  hipMemsetAsync(meta, 0, 64, stream);  // cnt + cursor

  k_cvt_x<<<(TT * HDIM / 8 + 255) / 256, 256, 0, stream>>>(x, xb, TT * HDIM / 8);

  const int rs = HDIM * IDIM / 16;        // runs per 1024x1024 matrix
  k_swz<<<(rs + 255) / 256, 256, 0, stream>>>(shg, wgs, HDIM, IDIM, 1);
  k_swz<<<(rs + 255) / 256, 256, 0, stream>>>(shu, wus, HDIM, IDIM, 1);
  k_swz<<<(rs + 255) / 256, 256, 0, stream>>>(shd, wds, IDIM, HDIM, 1);
  k_swz<<<(NE * rs + 255) / 256, 256, 0, stream>>>(exg, wge, HDIM, IDIM, NE);
  k_swz<<<(NE * rs + 255) / 256, 256, 0, stream>>>(exu, wue, HDIM, IDIM, NE);
  k_swz<<<(NE * rs + 255) / 256, 256, 0, stream>>>(exd, wde, IDIM, HDIM, NE);

  k_router<<<TT / 4, 128, 0, stream>>>(x, rw, tIdx, tW, meta);
  k_offsets<<<1, 32, 0, stream>>>(meta);
  k_assign<<<(TT + 255) / 256, 256, 0, stream>>>(tIdx, tW, meta, slotDest, slotW);

  const int mR = NSLOT / 128 + NE;  // 72 M-tiles covers worst-case padding
  k_gateup<false><<<dim3(TT / 128, IDIM / 64), 128, 0, stream>>>(
      xb, wgs, wus, hidS, nullptr, nullptr);
  k_gateup<true><<<dim3(mR, IDIM / 64), 128, 0, stream>>>(
      xb, wge, wue, hidR, meta, slotDest);

  k_down<false><<<dim3(TT / 128, HDIM / 64), 128, 0, stream>>>(
      hidS, wds, out, x, nullptr, nullptr, nullptr);
  k_down<true><<<dim3(mR, HDIM / 64), 128, 0, stream>>>(
      hidR, wde, sel, nullptr, meta, slotDest, slotW);

  k_combine<<<(TT * HDIM / 4 + 255) / 256, 256, 0, stream>>>(out, sel);
}
